// WindowAttention_30863634989263
// MI455X (gfx1250) — matrix operand order repaired and checked by reading, not compiled, not run on hardware
//
#include <hip/hip_runtime.h>
#include <hip/hip_bf16.h>
#include <math.h>

#define DIM   192
#define HEADS 6
#define HD    32
#define NTOK  49
#define NPAD  64
#define BW    4096

typedef _Float16 half_t;
typedef __attribute__((ext_vector_type(16))) _Float16 v16h;
typedef __attribute__((ext_vector_type(8)))  _Float16 v8h;
typedef __attribute__((ext_vector_type(8)))  float    v8f;

// ---------------- LDS layout (bytes) ----------------
#define OFF_QG   0          // [6][64][32] half = 24576   (aliased by O [64][192] half later)
#define OFF_KG   24576      // [6][64][32] half = 24576
#define OFF_VT   49152      // [6][32][64] half = 24576   (v transposed, zero padded)
#define OFF_P    73728      // [6][64][64] half = 49152   (per-head slot doubles as f32 gating scratch)
#define OFF_CA   122880     // [6][32]  f32
#define OFF_SAIN 123648     // [6][2][49] f32
#define OFF_SA   126000     // [6][49] f32
#define SMEM_BYTES 127232
#define OFF_O    0          // alias over QG region: [64][192] half = 24576

#define NEG_BIG  (-1.0e30f)

__device__ __forceinline__ float gelu_tanh(float x) {
  float x3 = x * x * x;
  return 0.5f * x * (1.0f + tanhf(0.7978845608028654f * (x + 0.044715f * x3)));
}
__device__ __forceinline__ float sigmoid_(float x) { return 1.0f / (1.0f + __expf(-x)); }

__device__ __forceinline__ v16h mk16(v8h lo, v8h hi) {
  v16h r;
#pragma unroll
  for (int i = 0; i < 8; ++i) { r[i] = lo[i]; r[i + 8] = hi[i]; }
  return r;
}
// A-fragment: 16 contiguous halves (elements 0..7 at p, 8..15 at p+8)
__device__ __forceinline__ v16h ldA16(const half_t* p) {
  return mk16(*(const v8h*)p, *(const v8h*)(p + 8));
}
// B-fragment: 16 contiguous halves
__device__ __forceinline__ v16h ldB16(const half_t* p) {
  return mk16(*(const v8h*)p, *(const v8h*)(p + 8));
}

// ========================================================================
// Kernel 1: PADDED relative position bias -> ws[6][64][64] f32.
//   rows >= 49 clone row 48 (keeps everything finite; those rows are unused)
//   cols >= 49 = -1e30   (softmax mask for padded keys; QK part there is 0)
// bias[n,m,h] = gelu(rel @ meta_w1 + b1) @ meta_w2 + b2 ;
// rel_d = pos_scale*(pef[n,d]-pef[m,d]) = scale*(pw[d,0]*dx + pw[d,1]*dy)
// ========================================================================
__global__ void bias_kernel(const float* __restrict__ pos_w,
                            const float* __restrict__ pos_scale,
                            const float* __restrict__ w1, const float* __restrict__ b1,
                            const float* __restrict__ w2, const float* __restrict__ b2,
                            float* __restrict__ biasOut) {
  int idx = blockIdx.x * blockDim.x + threadIdx.x;
  if (idx >= NPAD * NPAD) return;
  int n = idx / NPAD, m = idx % NPAD;
  float o[HEADS];
  if (m < NTOK) {
    int nc = (n < NTOK) ? n : (NTOK - 1);
    float scale = pos_scale[0];
    int ny = nc / 7, nx = nc % 7, my = m / 7, mx = m % 7;
    float dx = (float)(nx - mx) * (1.0f / 3.0f);  // linspace(-1,1,7): step 1/3
    float dy = (float)(ny - my) * (1.0f / 3.0f);
    float hacc[128];
#pragma unroll 4
    for (int j = 0; j < 128; ++j) hacc[j] = b1[j];
    for (int d = 0; d < DIM; ++d) {
      float rel = scale * (pos_w[d * 2 + 0] * dx + pos_w[d * 2 + 1] * dy);
#pragma unroll 4
      for (int j = 0; j < 128; ++j) hacc[j] += rel * w1[d * 128 + j];
    }
#pragma unroll
    for (int h = 0; h < HEADS; ++h) o[h] = b2[h];
    for (int j = 0; j < 128; ++j) {
      float g = gelu_tanh(hacc[j]);
#pragma unroll
      for (int h = 0; h < HEADS; ++h) o[h] += g * w2[j * HEADS + h];
    }
  } else {
#pragma unroll
    for (int h = 0; h < HEADS; ++h) o[h] = NEG_BIG;
  }
#pragma unroll
  for (int h = 0; h < HEADS; ++h)
    biasOut[h * NPAD * NPAD + n * NPAD + m] = o[h];
}

// ========================================================================
// PDSCA gating for one head's q or k block (one wave, lane = channel c).
// Stages the 49x32 f32 block into LDS scratch once (coalesced), then all
// gating math reads LDS. View: M[c,s] = flat[c*49+s], flat = (n,d) block.
// Writes gated values * outScale as f16 [64][32] (rows >= 49 zeroed).
// ========================================================================
__device__ __forceinline__ void pdsca_gate(
    const float* __restrict__ win, int colOff, float outScale,
    const float* __restrict__ caw1, const float* __restrict__ cab1,
    const float* __restrict__ caw2, const float* __restrict__ cab2,
    const float* __restrict__ saw,  const float* __restrict__ sab,
    half_t* __restrict__ dst, float* __restrict__ scratch,
    float* __restrict__ caS, float* __restrict__ sainS, float* __restrict__ saS,
    int lane) {
  // stage block to LDS (coalesced: lane = d)
  for (int n = 0; n < NTOK; ++n)
    scratch[n * HD + lane] = win[n * 576 + colOff + lane];
  // row mean -> ca[c]   (lane = c; row is contiguous 49 floats)
  float rs = 0.0f;
  for (int s = 0; s < NTOK; ++s) rs += scratch[lane * NTOK + s];
  caS[lane] = rs * (1.0f / 49.0f);
  // column mean / max: lane handles spatial position(s) s, loops channels
  for (int s = lane; s < NTOK; s += 32) {
    float cs = 0.0f, cm = -3.0e38f;
    for (int c = 0; c < HD; ++c) {
      float v = scratch[c * NTOK + s];
      cs += v;
      cm = fmaxf(cm, v);
    }
    sainS[s] = cs * (1.0f / 32.0f);
    sainS[NTOK + s] = cm;
  }
  // channel MLP: ca2[c] = b2[c] + sum_j w2[c,j]*gelu(b1[j] + sum_c' w1[j,c']*ca[c'])
  float ca2 = cab2[lane];
#pragma unroll
  for (int j = 0; j < 8; ++j) {
    float a = cab1[j];
#pragma unroll 8
    for (int c2 = 0; c2 < 32; ++c2) a += caw1[j * 32 + c2] * caS[c2];
    ca2 += caw2[lane * 8 + j] * gelu_tanh(a);
  }
  // 7x7 SAME conv on (2,7,7) spatial stats
  for (int s = lane; s < NTOK; s += 32) {
    int y = s / 7, x = s % 7;
    float acc = sab[0];
    for (int i = 0; i < 7; ++i) {
      int yy = y + i - 3;
      if ((unsigned)yy < 7u)
        for (int j = 0; j < 7; ++j) {
          int xx = x + j - 3;
          if ((unsigned)xx < 7u) {
            int sp = yy * 7 + xx;
            acc += saw[i * 7 + j] * sainS[sp] + saw[NTOK + i * 7 + j] * sainS[NTOK + sp];
          }
        }
    }
    saS[s] = acc;
  }
  // gate + store f16; flat index f = c*49+s is also the (n,d) flat index
  for (int s = 0; s < NTOK; ++s) {
    float g = sigmoid_(ca2 + saS[s]);
    int f = lane * NTOK + s;
    dst[f] = (half_t)(scratch[f] * g * outScale);
  }
  for (int n = NTOK; n < NPAD; ++n) dst[n * HD + lane] = (half_t)0.0f;
}

// ========================================================================
// Kernel 2: full window attention. One block = one window, 6 waves = 6 heads.
// ========================================================================
__global__ void __launch_bounds__(192)
attn_kernel(const float* __restrict__ qkv, const float* __restrict__ log_temp,
            const float* __restrict__ q_ca_w1, const float* __restrict__ q_ca_b1,
            const float* __restrict__ q_ca_w2, const float* __restrict__ q_ca_b2,
            const float* __restrict__ q_sa_w,  const float* __restrict__ q_sa_b,
            const float* __restrict__ k_ca_w1, const float* __restrict__ k_ca_b1,
            const float* __restrict__ k_ca_w2, const float* __restrict__ k_ca_b2,
            const float* __restrict__ k_sa_w,  const float* __restrict__ k_sa_b,
            const float* __restrict__ proj_w,  const float* __restrict__ biasPad,
            float* __restrict__ out) {
  extern __shared__ __align__(16) char smem[];
  half_t* QG = (half_t*)(smem + OFF_QG);
  half_t* KG = (half_t*)(smem + OFF_KG);
  half_t* VT = (half_t*)(smem + OFF_VT);
  half_t* PM = (half_t*)(smem + OFF_P);
  half_t* OM = (half_t*)(smem + OFF_O);
  float*  CAS  = (float*)(smem + OFF_CA);
  float*  SAIN = (float*)(smem + OFF_SAIN);
  float*  SAS  = (float*)(smem + OFF_SA);

  const int b    = blockIdx.x;
  const int h    = threadIdx.x >> 5;
  const int lane = threadIdx.x & 31;
  const float* win = qkv + (size_t)b * (NTOK * 3 * DIM);
  const float temperature = __expf(log_temp[0]);   // folded into gated q

  half_t* qg = QG + h * NPAD * HD;
  half_t* kg = KG + h * NPAD * HD;
  half_t* vt = VT + h * HD * NPAD;
  half_t* pm = PM + h * NPAD * NPAD;
  float*  gscratch = (float*)pm;                   // 6272 B <= 8192 B per-head P slot

  // ---- Phase A: gating + staging (wave-local, no barriers needed) ----
  pdsca_gate(win, 0 * DIM + h * HD, temperature,
             q_ca_w1, q_ca_b1, q_ca_w2, q_ca_b2, q_sa_w, q_sa_b,
             qg, gscratch, CAS + h * 32, SAIN + h * 98, SAS + h * NTOK, lane);
  pdsca_gate(win, 1 * DIM + h * HD, 1.0f,
             k_ca_w1, k_ca_b1, k_ca_w2, k_ca_b2, k_sa_w, k_sa_b,
             kg, gscratch, CAS + h * 32, SAIN + h * 98, SAS + h * NTOK, lane);
  { // v: transpose-copy into [32][64] f16, zero padded
    for (int n = 0; n < NTOK; ++n)
      vt[lane * NPAD + n] = (half_t)win[n * 576 + 2 * DIM + h * HD + lane];
    for (int n = NTOK; n < NPAD; ++n) vt[lane * NPAD + n] = (half_t)0.0f;
  }

  const int col = lane & 15;
  const int hs  = (lane < 16) ? 0 : 1;       // half-wave selector
  const v8f zc  = {0.f, 0.f, 0.f, 0.f, 0.f, 0.f, 0.f, 0.f};

  // ---- Phase B: S = (q*temp) k^T + bias (via WMMA C-init), softmax -> P ----
  const float* bph = biasPad + h * NPAD * NPAD;
  v16h bk[4];
#pragma unroll
  for (int tj = 0; tj < 4; ++tj)
    bk[tj] = ldB16(kg + (tj * 16 + col) * HD + hs * 16);

#pragma unroll
  for (int ti = 0; ti < 4; ++ti) {
    const int rowb = ti * 16 + hs * 8;
    v16h aq = ldA16(qg + (ti * 16 + col) * HD + hs * 16);
    v8f acc[4];
#pragma unroll
    for (int tj = 0; tj < 4; ++tj) {
      v8f cf;
#pragma unroll
      for (int r = 0; r < 8; ++r)
        cf[r] = bph[(rowb + r) * NPAD + tj * 16 + col];   // unconditional, coalesced
      acc[tj] = __builtin_amdgcn_wmma_f32_16x16x32_f16(false, aq, false, bk[tj],
                                                       (short)0, cf, false, false);
    }
    float rmx[8], rsm[8];
#pragma unroll
    for (int r = 0; r < 8; ++r) {
      float mx = fmaxf(fmaxf(acc[0][r], acc[1][r]), fmaxf(acc[2][r], acc[3][r]));
#pragma unroll
      for (int off = 1; off <= 8; off <<= 1) mx = fmaxf(mx, __shfl_xor(mx, off));
      rmx[r] = mx;
      rsm[r] = 0.0f;
    }
#pragma unroll
    for (int tj = 0; tj < 4; ++tj)
#pragma unroll
      for (int r = 0; r < 8; ++r) {
        float e = __expf(acc[tj][r] - rmx[r]);
        acc[tj][r] = e;
        rsm[r] += e;
      }
#pragma unroll
    for (int r = 0; r < 8; ++r) {
      float s = rsm[r];
#pragma unroll
      for (int off = 1; off <= 8; off <<= 1) s += __shfl_xor(s, off);
      rsm[r] = 1.0f / s;
    }
#pragma unroll
    for (int tj = 0; tj < 4; ++tj)
#pragma unroll
      for (int r = 0; r < 8; ++r)
        pm[(rowb + r) * NPAD + tj * 16 + col] = (half_t)(acc[tj][r] * rsm[r]);
  }

  __syncthreads();   // all waves done reading qg/kg; O may now alias qg

  // ---- Phase C: O = P @ V  (K = 64 in 2 steps), store f16 [64][192] ----
  v16h bv[2][2];
#pragma unroll
  for (int ks = 0; ks < 2; ++ks)
#pragma unroll
    for (int cI = 0; cI < 2; ++cI)
      bv[ks][cI] = ldB16(vt + (cI * 16 + col) * NPAD + ks * 32 + hs * 16);

#pragma unroll
  for (int ti = 0; ti < 4; ++ti) {
    v8f o0 = zc, o1 = zc;
#pragma unroll
    for (int ks = 0; ks < 2; ++ks) {
      v16h ap = ldA16(pm + (ti * 16 + col) * NPAD + ks * 32 + hs * 16);
      o0 = __builtin_amdgcn_wmma_f32_16x16x32_f16(false, ap, false, bv[ks][0],
                                                  (short)0, o0, false, false);
      o1 = __builtin_amdgcn_wmma_f32_16x16x32_f16(false, ap, false, bv[ks][1],
                                                  (short)0, o1, false, false);
    }
#pragma unroll
    for (int r = 0; r < 8; ++r) {
      int row = ti * 16 + hs * 8 + r;
      OM[row * DIM + h * HD + col]      = (half_t)o0[r];
      OM[row * DIM + h * HD + 16 + col] = (half_t)o1[r];
    }
  }

  __syncthreads();   // O fully written, visible to all waves

  // ---- Phase D: OUT = O @ proj_w^T  (M=64, N=192, K=192) ----
#pragma unroll
  for (int njI = 0; njI < 2; ++njI) {
    const int nj = h + njI * 6;
    v16h bw[6];
#pragma unroll
    for (int ks = 0; ks < 6; ++ks) {
      const float* pw = proj_w + (nj * 16 + col) * DIM + ks * 32 + hs * 16;
      v16h t;
#pragma unroll
      for (int e = 0; e < 16; ++e) t[e] = (half_t)pw[e];
      bw[ks] = t;
    }
#pragma unroll
    for (int ti = 0; ti < 4; ++ti) {
      v8f accp = zc;
#pragma unroll
      for (int ks = 0; ks < 6; ++ks) {
        v16h ao = ldA16(OM + (ti * 16 + col) * DIM + ks * 32 + hs * 16);
        accp = __builtin_amdgcn_wmma_f32_16x16x32_f16(false, ao, false, bw[ks],
                                                      (short)0, accp, false, false);
      }
#pragma unroll
      for (int r = 0; r < 8; ++r) {
        int row = ti * 16 + hs * 8 + r;
        if (row < NTOK)
          out[(size_t)b * NTOK * DIM + row * DIM + nj * 16 + col] = accp[r];
      }
    }
  }
}

// ========================================================================
extern "C" void kernel_launch(void* const* d_in, const int* in_sizes, int n_in,
                              void* d_out, int out_size, void* d_ws, size_t ws_size,
                              hipStream_t stream) {
  const float* qkv       = (const float*)d_in[0];
  const float* log_temp  = (const float*)d_in[1];
  const float* pos_w     = (const float*)d_in[2];
  // d_in[3] = pos_b (cancels in pairwise diff)
  const float* pos_scale = (const float*)d_in[4];
  const float* meta_w1   = (const float*)d_in[5];
  const float* meta_b1   = (const float*)d_in[6];
  const float* meta_w2   = (const float*)d_in[7];
  const float* meta_b2   = (const float*)d_in[8];
  const float* q_ca_w1   = (const float*)d_in[9];
  const float* q_ca_b1   = (const float*)d_in[10];
  const float* q_ca_w2   = (const float*)d_in[11];
  const float* q_ca_b2   = (const float*)d_in[12];
  const float* q_sa_w    = (const float*)d_in[13];
  const float* q_sa_b    = (const float*)d_in[14];
  const float* k_ca_w1   = (const float*)d_in[15];
  const float* k_ca_b1   = (const float*)d_in[16];
  const float* k_ca_w2   = (const float*)d_in[17];
  const float* k_ca_b2   = (const float*)d_in[18];
  const float* k_sa_w    = (const float*)d_in[19];
  const float* k_sa_b    = (const float*)d_in[20];
  const float* proj_w    = (const float*)d_in[21];
  float* bias = (float*)d_ws;   // padded [6][64][64] f32 = 98,304 B

  (void)hipFuncSetAttribute((const void*)attn_kernel,
                            hipFuncAttributeMaxDynamicSharedMemorySize, SMEM_BYTES);

  bias_kernel<<<(NPAD * NPAD + 255) / 256, 256, 0, stream>>>(
      pos_w, pos_scale, meta_w1, meta_b1, meta_w2, meta_b2, bias);

  attn_kernel<<<BW, 192, SMEM_BYTES, stream>>>(
      qkv, log_temp,
      q_ca_w1, q_ca_b1, q_ca_w2, q_ca_b2, q_sa_w, q_sa_b,
      k_ca_w1, k_ca_b1, k_ca_w2, k_ca_b2, k_sa_w, k_sa_b,
      proj_w, bias, (float*)d_out);
}
